// GINEPretrainBlock_37486474559540
// MI455X (gfx1250) — matrix operand order repaired and checked by reading, not compiled, not run on hardware
//
#include <hip/hip_runtime.h>

#define N_NODES 50000
#define N_EDGES 640000
#define D       128
#define LN_EPS  1e-5f
#define WPITCH  136   // f16 pitch for transposed weights in LDS (272B rows -> conflict-free)
#define HPITCH  136   // f16 pitch for per-wave activation tile in LDS

typedef __attribute__((ext_vector_type(16))) _Float16 v16h;
typedef __attribute__((ext_vector_type(8)))  _Float16 v8h;
typedef __attribute__((ext_vector_type(4)))  _Float16 v4h;
typedef __attribute__((ext_vector_type(8)))  float    v8f;

// ---------------------------------------------------------------------------
// Kernel 1: msg = relu(x[src] + edge_attr); atomic scatter-add into agg[dst].
// One wave per edge (src/dst wave-uniform), float4 per lane = 512B coalesced.
// ---------------------------------------------------------------------------
__global__ __launch_bounds__(256) void gine_edge_kernel(
    const float* __restrict__ x, const long long* __restrict__ ei,
    const float* __restrict__ ea, float* __restrict__ agg)
{
    unsigned tid = blockIdx.x * 256u + threadIdx.x;
    unsigned e = tid >> 5;
    unsigned c = tid & 31;
    if (e >= N_EDGES) return;
    long long s = ei[e];
    long long d = ei[N_EDGES + e];
    float4 xv = ((const float4*)(x + s * (long long)D))[c];
    float4 ev = ((const float4*)(ea + (long long)e * D))[c];
    float4 m;
    m.x = fmaxf(xv.x + ev.x, 0.0f);
    m.y = fmaxf(xv.y + ev.y, 0.0f);
    m.z = fmaxf(xv.z + ev.z, 0.0f);
    m.w = fmaxf(xv.w + ev.w, 0.0f);
    float* p = agg + d * (long long)D + c * 4;
    atomicAdd(p + 0, m.x);
    atomicAdd(p + 1, m.y);
    atomicAdd(p + 2, m.z);
    atomicAdd(p + 3, m.w);
}

// ---------------------------------------------------------------------------
// 16x128 @ 128x128 GEMM tile on one wave via v_wmma_f32_16x16x32_f16.
// A: 16 rows of f16 in LDS (pitch HPITCH). Bt: transposed weights Bt[n][k]
// f16 in LDS (pitch WPITCH). acc[t] = 16x16 f32 tile for out cols [16t,16t+16).
// ---------------------------------------------------------------------------
__device__ __forceinline__ void gemm_tile(const _Float16* A, const _Float16* Bt,
                                          const float* __restrict__ bias,
                                          int l16, int hi, v8f acc[8])
{
#pragma unroll
    for (int t = 0; t < 8; ++t) {
        float bv = bias[t * 16 + l16];
        v8f c = {bv, bv, bv, bv, bv, bv, bv, bv};
        acc[t] = c;
    }
#pragma unroll
    for (int ks = 0; ks < 4; ++ks) {
        const int kbase = ks * 32;
        // A fragment (ISA 16-bit A layout): lane half 'hi' holds K = kbase+8*hi..+7
        // in VGPR0-3 and K = kbase+16+8*hi..+7 in VGPR4-7.
        const _Float16* ap = A + l16 * HPITCH + kbase + 8 * hi;
        v8h a0 = *(const v8h*)(ap);
        v8h a1 = *(const v8h*)(ap + 16);
        v16h a = __builtin_shufflevector(a0, a1, 0, 1, 2, 3, 4, 5, 6, 7,
                                         8, 9, 10, 11, 12, 13, 14, 15);
#pragma unroll
        for (int t = 0; t < 8; ++t) {
            // B fragment: lane = out-col n = 16t+l16; K = kbase+8*hi..+7 and
            // kbase+16+8*hi..+7 (same order as A) in transposed weights.
            const _Float16* bp = Bt + (t * 16 + l16) * WPITCH + kbase + 8 * hi;
            v8h b0 = *(const v8h*)(bp);
            v8h b1 = *(const v8h*)(bp + 16);
            v16h b = __builtin_shufflevector(b0, b1, 0, 1, 2, 3, 4, 5, 6, 7,
                                             8, 9, 10, 11, 12, 13, 14, 15);
            acc[t] = __builtin_amdgcn_wmma_f32_16x16x32_f16(
                false, a, false, b, (short)0, acc[t], false, false);
        }
    }
}

// ---------------------------------------------------------------------------
// Kernel 2: h = (1+eps)*x + agg; h = relu(h@W1+b1)@W2+b2; LayerNorm; ReLU.
// 8 waves/WG, 16 rows per wave, f16 operands, f32 WMMA accumulation.
// ---------------------------------------------------------------------------
__global__ __launch_bounds__(256) void gine_mlp_kernel(
    const float* __restrict__ x, const float* __restrict__ agg,
    const float* __restrict__ W1, const float* __restrict__ b1,
    const float* __restrict__ W2, const float* __restrict__ b2,
    const float* __restrict__ epsp, const float* __restrict__ gamma,
    const float* __restrict__ beta, float* __restrict__ out)
{
    extern __shared__ _Float16 smem[];
    _Float16* sW1t = smem;                       // 128 * WPITCH
    _Float16* sW2t = sW1t + 128 * WPITCH;        // 128 * WPITCH
    _Float16* sH   = sW2t + 128 * WPITCH;        // 8 waves * 16 * HPITCH

    const int tid  = threadIdx.x;
    const int wave = tid >> 5;
    const int lane = tid & 31;
    const int l16  = lane & 15;
    const int hi   = lane >> 4;

    // Stage both weight matrices transposed + converted to f16 (coalesced
    // global reads; LDS writes stride 272B -> conflict-free).
    for (int i = tid; i < D * D; i += 256) {
        int k = i >> 7, n = i & 127;
        sW1t[n * WPITCH + k] = (_Float16)W1[i];
        sW2t[n * WPITCH + k] = (_Float16)W2[i];
    }
    __syncthreads();

    const float eps1 = 1.0f + epsp[0];
    const int rowBase = blockIdx.x * 128 + wave * 16;
    _Float16* hs = sH + wave * (16 * HPITCH);

    // h_in = (1+eps)*x + agg, staged per-wave as f16 (rows clamped for tail).
#pragma unroll 4
    for (int r = 0; r < 16; ++r) {
        int row = rowBase + r;
        int rc = row < N_NODES ? row : N_NODES - 1;
        float4 xv = ((const float4*)(x + (size_t)rc * D))[lane];
        float4 av = ((const float4*)(agg + (size_t)rc * D))[lane];
        v4h h;
        h[0] = (_Float16)(eps1 * xv.x + av.x);
        h[1] = (_Float16)(eps1 * xv.y + av.y);
        h[2] = (_Float16)(eps1 * xv.z + av.z);
        h[3] = (_Float16)(eps1 * xv.w + av.w);
        *(v4h*)(hs + r * HPITCH + lane * 4) = h;
    }

    v8f acc[8];
    gemm_tile(hs, sW1t, b1, l16, hi, acc);   // GEMM1 (+b1)

    // ReLU, restage H1 as f16 into the same per-wave tile (D-layout:
    // acc[t][v] = H1[v+8*hi][16t+l16]; per-wave DS ops are in order).
#pragma unroll
    for (int t = 0; t < 8; ++t)
#pragma unroll
        for (int v = 0; v < 8; ++v)
            hs[(v + 8 * hi) * HPITCH + t * 16 + l16] =
                (_Float16)fmaxf(acc[t][v], 0.0f);

    gemm_tile(hs, sW2t, b2, l16, hi, acc);   // GEMM2 (+b2)

    // LayerNorm over the feature dim, fully in registers: row r = v+8*hi,
    // reduce across the 16-lane half (hi bit is never mixed).
    float mean[8], inv[8];
#pragma unroll
    for (int v = 0; v < 8; ++v) {
        float s = 0.0f, q = 0.0f;
#pragma unroll
        for (int t = 0; t < 8; ++t) { float val = acc[t][v]; s += val; q += val * val; }
#pragma unroll
        for (int m = 1; m < 16; m <<= 1) {
            s += __shfl_xor(s, m, 32);
            q += __shfl_xor(q, m, 32);
        }
        float mu  = s * (1.0f / D);
        float var = q * (1.0f / D) - mu * mu;
        mean[v] = mu;
        inv[v]  = rsqrtf(var + LN_EPS);
    }

#pragma unroll
    for (int t = 0; t < 8; ++t) {
        float g  = gamma[t * 16 + l16];
        float bb = beta[t * 16 + l16];
#pragma unroll
        for (int v = 0; v < 8; ++v) {
            int row = rowBase + v + 8 * hi;
            if (row < N_NODES) {
                float val = (acc[t][v] - mean[v]) * inv[v] * g + bb;
                out[(size_t)row * D + t * 16 + l16] = fmaxf(val, 0.0f);
            }
        }
    }
}

extern "C" void kernel_launch(void* const* d_in, const int* in_sizes, int n_in,
                              void* d_out, int out_size, void* d_ws, size_t ws_size,
                              hipStream_t stream) {
    (void)in_sizes; (void)n_in; (void)out_size; (void)ws_size;
    const float*     x     = (const float*)d_in[0];
    const long long* ei    = (const long long*)d_in[1];   // int64 edge_index [2,E]
    const float*     ea    = (const float*)d_in[2];
    const float*     W1    = (const float*)d_in[3];
    const float*     b1    = (const float*)d_in[4];
    const float*     W2    = (const float*)d_in[5];
    const float*     b2    = (const float*)d_in[6];
    const float*     eps   = (const float*)d_in[7];
    const float*     gamma = (const float*)d_in[8];
    const float*     beta  = (const float*)d_in[9];
    float* out = (float*)d_out;
    float* agg = (float*)d_ws;                            // N*D f32 scratch

    hipMemsetAsync(agg, 0, (size_t)N_NODES * D * sizeof(float), stream);

    const int edgeThreads = N_EDGES * 32;                 // one wave per edge
    gine_edge_kernel<<<(edgeThreads + 255) / 256, 256, 0, stream>>>(x, ei, ea, agg);

    const size_t smemBytes =
        (size_t)(2 * 128 * WPITCH + 8 * 16 * HPITCH) * sizeof(_Float16); // ~102 KB
    hipFuncSetAttribute(reinterpret_cast<const void*>(gine_mlp_kernel),
                        hipFuncAttributeMaxDynamicSharedMemorySize,
                        (int)smemBytes);
    gine_mlp_kernel<<<(N_NODES + 127) / 128, 256, smemBytes, stream>>>(
        x, agg, W1, b1, W2, b2, eps, gamma, beta, out);
}
